// IPFluxAttnProcessor2_0_50740743635744
// MI455X (gfx1250) — matrix operand order repaired and checked by reading, not compiled, not run on hardware
//
#include <hip/hip_runtime.h>
#include <hip/hip_bf16.h>

// ---------------------------------------------------------------------------
// IP-Flux attention for MI455X (gfx1250, wave32, WMMA 16x16x32 bf16)
//
//  1) 8x GEMM (Q/K/V, added Q/K/V, ip K/V)   -> gemm_bias_wmma (128x128 tile,
//     LDS double-buffered, float4 staging, v_perm bf16 packing, f32 accum)
//  2) per-head RMSNorm (q,k,eq,ek,ipk)       -> rmsnorm_kernel
//  3) IP attention (4 keys, VALU)            -> ip_attn_kernel
//  4) RoPE + seq-concat into (2560,3072)     -> rope_concat_kernel
//  5) flash attention w/ WMMA                -> flash_attn_wmma (128q block,
//     8 waves, V transposed in LDS, global_prefetch of next K/V block)
//  6) residual add of ip branch              -> add_scale_kernel
//  7) 2x output GEMM -> d_out                -> gemm_bias_wmma
// ---------------------------------------------------------------------------

typedef __attribute__((ext_vector_type(16))) __bf16 v16bf;
typedef __attribute__((ext_vector_type(8)))  float  v8f;

#define DEV static __device__ __forceinline__

DEV unsigned short f2us(float f) {            // f32 -> bf16 bits (round-half-up)
  unsigned u = __builtin_bit_cast(unsigned, f) + 0x8000u;
  return (unsigned short)(u >> 16);
}
DEV __bf16 us2bf(unsigned short u) { return __builtin_bit_cast(__bf16, u); }

DEV unsigned pack2_sw(float x, float y) {     // {bf16(y), bf16(x)} via v_perm
  unsigned ux = __builtin_bit_cast(unsigned, x) + 0x8000u;
  unsigned uy = __builtin_bit_cast(unsigned, y) + 0x8000u;
  // D.b0 = S1.b2, D.b1 = S1.b3, D.b2 = S0.b2, D.b3 = S0.b3
  return __builtin_amdgcn_perm(uy, ux, 0x07060302u);
}
template <class T>
DEV unsigned pk_bits_or_sw(T r, float x, float y) {
  if constexpr (sizeof(T) == 4) {
    return __builtin_bit_cast(unsigned, r);
  } else {
    return pack2_sw(x, y);
  }
}
DEV unsigned pack2(float x, float y) {        // 2x f32 -> packed bf16x2
#if __has_builtin(__builtin_amdgcn_cvt_pk_bf16_f32)
  return pk_bits_or_sw(__builtin_amdgcn_cvt_pk_bf16_f32(x, y), x, y);
#else
  return pack2_sw(x, y);
#endif
}
DEV void wavebar() {
#if __has_builtin(__builtin_amdgcn_wave_barrier)
  __builtin_amdgcn_wave_barrier();
#else
  __syncthreads();
#endif
}

// ---------------------------------------------------------------------------
// C[MxN] = A[MxK] * W[NxK]^T + bias   (f32 in/out, bf16 WMMA compute)
// 256 threads = 8 waves; tile 128x128, K-step 32, LDS double buffered.
// wave (wr = w>>1, wc = w&1): rows wr*32..+32, cols wc*64..+64
//   = 2 A-frags x 4 B-frags = 8 WMMAs per K-step.
// Requires: N % 128 == 0, K % 32 == 0 (true for all launches; M may ragged).
// ---------------------------------------------------------------------------
__global__ __launch_bounds__(256) void gemm_bias_wmma(
    const float* __restrict__ A, const float* __restrict__ W,
    const float* __restrict__ bias, float* __restrict__ C,
    int M, int N, int K)
{
  __shared__ __align__(16) unsigned short la[2][128][36];
  __shared__ __align__(16) unsigned short lb[2][128][36];

  const int tid  = threadIdx.x;
  const int lane = tid & 31, w = tid >> 5;
  const int g = lane >> 4, ln = lane & 15;
  const int wr = w >> 1, wc = w & 1;
  const int n0 = blockIdx.x * 128;
  const int m0 = blockIdx.y * 128;
  const bool full = (m0 + 128 <= M);

  v8f acc[2][4];
  #pragma unroll
  for (int t = 0; t < 2; ++t)
    #pragma unroll
    for (int c = 0; c < 4; ++c)
      #pragma unroll
      for (int r = 0; r < 8; ++r) acc[t][c][r] = 0.f;

  float4 ra[4], rb[4];        // register staging for next tile

  // 1024 float4 per matrix per tile; 256 thr x 4
  auto loadTile = [&](int kk) {
    #pragma unroll
    for (int i = 0; i < 4; ++i) {
      int f = tid + 256 * i;
      int row = f >> 3, c4 = f & 7;
      int col = kk + c4 * 4;
      int gm = m0 + row, gn = n0 + row;
      if (full) {
        ra[i] = *(const float4*)(A + (size_t)gm * K + col);
      } else {
        int gmc = gm < M ? gm : (M - 1);
        float4 t = *(const float4*)(A + (size_t)gmc * K + col);
        ra[i].x = gm < M ? t.x : 0.f;  ra[i].y = gm < M ? t.y : 0.f;
        ra[i].z = gm < M ? t.z : 0.f;  ra[i].w = gm < M ? t.w : 0.f;
      }
      rb[i] = *(const float4*)(W + (size_t)gn * K + col);   // N always full
    }
  };
  auto storeTile = [&](int buf) {
    #pragma unroll
    for (int i = 0; i < 4; ++i) {
      int f = tid + 256 * i;
      int row = f >> 3, c4 = f & 7;
      uint2 pa, pb;
      pa.x = pack2(ra[i].x, ra[i].y);  pa.y = pack2(ra[i].z, ra[i].w);
      pb.x = pack2(rb[i].x, rb[i].y);  pb.y = pack2(rb[i].z, rb[i].w);
      *(uint2*)&la[buf][row][c4 * 4] = pa;
      *(uint2*)&lb[buf][row][c4 * 4] = pb;
    }
  };
  auto compute = [&](int buf) {
    v16bf af[2];
    #pragma unroll
    for (int t = 0; t < 2; ++t)
      #pragma unroll
      for (int j = 0; j < 16; ++j)      // A-frag: k = j + 8g + (j&8)
        af[t][j] = us2bf(la[buf][wr * 32 + t * 16 + ln][j + 8 * g + (j & 8)]);
    #pragma unroll
    for (int c = 0; c < 4; ++c) {
      v16bf bf_;
      #pragma unroll
      for (int j = 0; j < 16; ++j)      // B-frag: k = j + 8g + (j&8)
        bf_[j] = us2bf(lb[buf][wc * 64 + c * 16 + ln][j + 8 * g + (j & 8)]);
      acc[0][c] = __builtin_amdgcn_wmma_f32_16x16x32_bf16(
          false, af[0], false, bf_, (short)0, acc[0][c], false, false);
      acc[1][c] = __builtin_amdgcn_wmma_f32_16x16x32_bf16(
          false, af[1], false, bf_, (short)0, acc[1][c], false, false);
    }
  };

  const int nT = K >> 5;                 // K % 32 == 0
  loadTile(0);
  storeTile(0);
  __syncthreads();
  int cur = 0;
  for (int t = 0; t < nT; ++t) {
    if (t + 1 < nT) loadTile((t + 1) * 32);   // global->regs overlaps WMMAs
    compute(cur);
    if (t + 1 < nT) storeTile(cur ^ 1);
    __syncthreads();
    cur ^= 1;
  }

  if (full) {
    #pragma unroll
    for (int t = 0; t < 2; ++t)
      #pragma unroll
      for (int c = 0; c < 4; ++c) {
        int col = n0 + wc * 64 + c * 16 + ln;
        #pragma unroll
        for (int r = 0; r < 8; ++r) {   // C/D: row = r + 8g
          int row = m0 + wr * 32 + t * 16 + 8 * g + r;
          float val = acc[t][c][r];
          if (bias) val += bias[col];
          C[(size_t)row * N + col] = val;
        }
      }
  } else {
    #pragma unroll
    for (int t = 0; t < 2; ++t)
      #pragma unroll
      for (int c = 0; c < 4; ++c) {
        int col = n0 + wc * 64 + c * 16 + ln;
        #pragma unroll
        for (int r = 0; r < 8; ++r) {
          int row = m0 + wr * 32 + t * 16 + 8 * g + r;
          if (row < M) {
            float val = acc[t][c][r];
            if (bias) val += bias[col];
            C[(size_t)row * N + col] = val;
          }
        }
      }
  }
}

// ---------------------------------------------------------------------------
// Per-(token,head) RMSNorm over 128 dims in place; one wave per row.
// ---------------------------------------------------------------------------
__global__ __launch_bounds__(256) void rmsnorm_kernel(
    float* __restrict__ x, const float* __restrict__ wgt, float eps, int rows)
{
  int wid = blockIdx.x * 8 + (threadIdx.x >> 5);
  if (wid >= rows) return;
  int lane = threadIdx.x & 31;
  float* p = x + (size_t)wid * 128;

  float v[4]; float ss = 0.f;
  #pragma unroll
  for (int i = 0; i < 4; ++i) { v[i] = p[lane + 32 * i]; ss += v[i] * v[i]; }
  #pragma unroll
  for (int off = 1; off < 32; off <<= 1) ss += __shfl_xor(ss, off, 32);
  float inv = rsqrtf(ss * (1.0f / 128.0f) + eps);
  #pragma unroll
  for (int i = 0; i < 4; ++i) {
    float y = v[i] * inv;
    if (wgt) y *= wgt[lane + 32 * i];
    p[lane + 32 * i] = y;
  }
}

// ---------------------------------------------------------------------------
// RoPE + concat: out(2560,3072); s<512 from enc buf else img buf (s-512).
// ---------------------------------------------------------------------------
__global__ __launch_bounds__(256) void rope_concat_kernel(
    const float* __restrict__ enc, const float* __restrict__ img,
    const float* __restrict__ cosb, const float* __restrict__ sinb,
    float* __restrict__ out, int do_rope)
{
  int idx = blockIdx.x * blockDim.x + threadIdx.x;
  const int total = 2560 * 1536;          // dim pairs
  if (idx >= total) return;
  int s   = idx / 1536;
  int rem = idx - s * 1536;
  int h   = rem >> 6;
  int dp  = rem & 63;
  int d0  = 2 * dp, d1 = d0 + 1;
  const float* src = (s < 512) ? (enc + (size_t)s * 3072)
                               : (img + (size_t)(s - 512) * 3072);
  float x0 = src[h * 128 + d0];
  float x1 = src[h * 128 + d1];
  float o0 = x0, o1 = x1;
  if (do_rope) {
    float c0 = cosb[s * 128 + d0], sn0 = sinb[s * 128 + d0];
    float c1 = cosb[s * 128 + d1], sn1 = sinb[s * 128 + d1];
    o0 = x0 * c0 - x1 * sn0;
    o1 = x1 * c1 + x0 * sn1;
  }
  out[(size_t)s * 3072 + h * 128 + d0] = o0;
  out[(size_t)s * 3072 + h * 128 + d1] = o1;
}

// ---------------------------------------------------------------------------
// IP attention: q(2048,24,128) vs ip_k/ip_v(4,24,128). One wave per (s,h).
// ---------------------------------------------------------------------------
__global__ __launch_bounds__(256) void ip_attn_kernel(
    const float* __restrict__ q, const float* __restrict__ ipk,
    const float* __restrict__ ipv, float* __restrict__ out)
{
  int wid = blockIdx.x * 8 + (threadIdx.x >> 5);
  if (wid >= 2048 * 24) return;
  int lane = threadIdx.x & 31;
  int s = wid / 24, h = wid - s * 24;

  const float* qp = q + (size_t)s * 3072 + h * 128;
  float qv[4];
  #pragma unroll
  for (int i = 0; i < 4; ++i) qv[i] = qp[lane + 32 * i];

  float sc[4];
  #pragma unroll
  for (int j = 0; j < 4; ++j) {
    const float* kp = ipk + (size_t)j * 3072 + h * 128;
    float d = 0.f;
    #pragma unroll
    for (int i = 0; i < 4; ++i) d += qv[i] * kp[lane + 32 * i];
    #pragma unroll
    for (int off = 1; off < 32; off <<= 1) d += __shfl_xor(d, off, 32);
    sc[j] = d * 0.08838834764831845f;     // 1/sqrt(128)
  }
  float mx = fmaxf(fmaxf(sc[0], sc[1]), fmaxf(sc[2], sc[3]));
  float e[4], sum = 0.f;
  #pragma unroll
  for (int j = 0; j < 4; ++j) { e[j] = __expf(sc[j] - mx); sum += e[j]; }
  float o[4] = {0.f, 0.f, 0.f, 0.f};
  #pragma unroll
  for (int j = 0; j < 4; ++j) {
    const float* vp = ipv + (size_t)j * 3072 + h * 128;
    float pj = e[j] / sum;
    #pragma unroll
    for (int i = 0; i < 4; ++i) o[i] += pj * vp[lane + 32 * i];
  }
  float* op = out + (size_t)s * 3072 + h * 128;
  #pragma unroll
  for (int i = 0; i < 4; ++i) op[lane + 32 * i] = o[i];
}

// ---------------------------------------------------------------------------
// Flash attention with WMMA. q/k/v/out: (2560, 3072) = (S, 24 heads * 128).
// grid = 24 heads * 20 query-blocks of 128; block = 256 thr (8 waves,
// 16 q-rows each). Key blocks of 32 staged to LDS bf16; V stored transposed
// so the PV B-fragment gather is contiguous. 16 WMMAs per wave per key block.
// P scratch is wave-private: wave_barrier (not block barrier) between
// P-store and P-gather; LDS ops within a wave are in-order (DScnt).
// ---------------------------------------------------------------------------
__global__ __launch_bounds__(256) void flash_attn_wmma(
    const float* __restrict__ q, const float* __restrict__ k,
    const float* __restrict__ v, float* __restrict__ out)
{
  __shared__ __align__(16) unsigned short lq[128][132];
  __shared__ __align__(16) unsigned short lk[32][132];
  __shared__ __align__(16) unsigned short lvT[128][36];   // [d][key]
  __shared__ unsigned short lp[8][16][40];  // per-wave P (16q x 32k)

  const int S = 2560, DS = 3072;
  const int tid  = threadIdx.x;
  const int lane = tid & 31, w = tid >> 5;
  const int g = lane >> 4, ln = lane & 15;
  const int h  = blockIdx.x / 20;
  const int qb = blockIdx.x % 20;
  const int q0 = qb * 128;

  // stage Q tile (128x128) -> bf16 LDS, float4 loads, packed b64 LDS stores
  #pragma unroll
  for (int i = 0; i < 16; ++i) {
    int f = tid + 256 * i;                 // 4096 float4 = 128 x 32
    int row = f >> 5, c4 = f & 31;
    float4 t = *(const float4*)(q + (size_t)(q0 + row) * DS + h * 128 + c4 * 4);
    uint2 p; p.x = pack2(t.x, t.y); p.y = pack2(t.z, t.w);
    *(uint2*)&lq[row][c4 * 4] = p;
  }
  __syncthreads();

  // Q fragments (4 d-chunks of 32) in registers for the whole kernel
  v16bf qf[4];
  #pragma unroll
  for (int dc = 0; dc < 4; ++dc)
    #pragma unroll
    for (int j = 0; j < 16; ++j)
      qf[dc][j] = us2bf(lq[w * 16 + ln][dc * 32 + j + 8 * g + (j & 8)]);

  float mi[8], li[8];
  v8f ao[8];
  #pragma unroll
  for (int r = 0; r < 8; ++r) { mi[r] = -1e30f; li[r] = 0.f; }
  #pragma unroll
  for (int dc = 0; dc < 8; ++dc)
    #pragma unroll
    for (int r = 0; r < 8; ++r) ao[dc][r] = 0.f;

  for (int kk = 0; kk < S; kk += 32) {
    // prefetch next K/V block into GL2 (global_prefetch_b8)
    if (kk + 32 < S) {
      int prow = kk + 32 + (tid >> 3), pcol = (tid & 7) * 16;
      __builtin_prefetch(k + (size_t)prow * DS + h * 128 + pcol, 0, 1);
      __builtin_prefetch(v + (size_t)prow * DS + h * 128 + pcol, 0, 1);
    }
    __syncthreads();                       // lk/lvT WAR vs prior PV reads
    #pragma unroll
    for (int i = 0; i < 4; ++i) {          // 1024 float4 = 32 rows x 32
      int f = tid + 256 * i;
      int row = f >> 5, c4 = f & 31;
      float4 tk = *(const float4*)(k + (size_t)(kk + row) * DS + h * 128 + c4 * 4);
      uint2 p; p.x = pack2(tk.x, tk.y); p.y = pack2(tk.z, tk.w);
      *(uint2*)&lk[row][c4 * 4] = p;
      float4 tv = *(const float4*)(v + (size_t)(kk + row) * DS + h * 128 + c4 * 4);
      lvT[c4 * 4 + 0][row] = f2us(tv.x);   // transposed store [d][key]
      lvT[c4 * 4 + 1][row] = f2us(tv.y);
      lvT[c4 * 4 + 2][row] = f2us(tv.z);
      lvT[c4 * 4 + 3][row] = f2us(tv.w);
    }
    __syncthreads();

    // S = Q*K^T : two 16-key tiles, contract over d in 4 chunks of 32
    v8f s0, s1;
    #pragma unroll
    for (int r = 0; r < 8; ++r) { s0[r] = 0.f; s1[r] = 0.f; }
    #pragma unroll
    for (int dc = 0; dc < 4; ++dc) {
      v16bf b0, b1;
      #pragma unroll
      for (int j = 0; j < 16; ++j) {
        int dloc = dc * 32 + j + 8 * g + (j & 8);
        b0[j] = us2bf(lk[ln][dloc]);
        b1[j] = us2bf(lk[16 + ln][dloc]);
      }
      s0 = __builtin_amdgcn_wmma_f32_16x16x32_bf16(
          false, qf[dc], false, b0, (short)0, s0, false, false);
      s1 = __builtin_amdgcn_wmma_f32_16x16x32_bf16(
          false, qf[dc], false, b1, (short)0, s1, false, false);
    }

    // online softmax (rows r+8g, 16 cols per lane-group)
    const float scale = 0.08838834764831845f;
    float mx[8], al[8], rs[8];
    #pragma unroll
    for (int r = 0; r < 8; ++r) {
      float a = s0[r] * scale, b = s1[r] * scale;
      s0[r] = a; s1[r] = b;
      mx[r] = fmaxf(a, b);
    }
    #pragma unroll
    for (int off = 1; off < 16; off <<= 1)
      #pragma unroll
      for (int r = 0; r < 8; ++r)
        mx[r] = fmaxf(mx[r], __shfl_xor(mx[r], off, 32));
    #pragma unroll
    for (int r = 0; r < 8; ++r) {
      float mnew = fmaxf(mi[r], mx[r]);
      al[r] = __expf(mi[r] - mnew);
      float p0 = __expf(s0[r] - mnew);
      float p1 = __expf(s1[r] - mnew);
      s0[r] = p0; s1[r] = p1;
      rs[r] = p0 + p1;
      mi[r] = mnew;
    }
    #pragma unroll
    for (int off = 1; off < 16; off <<= 1)
      #pragma unroll
      for (int r = 0; r < 8; ++r)
        rs[r] += __shfl_xor(rs[r], off, 32);
    #pragma unroll
    for (int r = 0; r < 8; ++r) li[r] = li[r] * al[r] + rs[r];
    #pragma unroll
    for (int dc = 0; dc < 8; ++dc)
      #pragma unroll
      for (int r = 0; r < 8; ++r) ao[dc][r] *= al[r];

    // P (C-layout f32) -> wave-private LDS bf16, re-gather as A-fragment
    #pragma unroll
    for (int r = 0; r < 8; ++r) {
      lp[w][8 * g + r][ln]      = f2us(s0[r]);
      lp[w][8 * g + r][16 + ln] = f2us(s1[r]);
    }
    wavebar();                             // same-wave LDS is in-order

    v16bf pf;
    #pragma unroll
    for (int j = 0; j < 16; ++j)
      pf[j] = us2bf(lp[w][ln][j + 8 * g + (j & 8)]);

    // O += P * V : 8 d-chunks of 16; vfrag contiguous thanks to lvT
    #pragma unroll
    for (int dc = 0; dc < 8; ++dc) {
      v16bf vf;
      #pragma unroll
      for (int j = 0; j < 16; ++j)
        vf[j] = us2bf(lvT[dc * 16 + ln][j + 8 * g + (j & 8)]);
      ao[dc] = __builtin_amdgcn_wmma_f32_16x16x32_bf16(
          false, pf, false, vf, (short)0, ao[dc], false, false);
    }
  }

  // epilogue: O / l
  #pragma unroll
  for (int dc = 0; dc < 8; ++dc)
    #pragma unroll
    for (int r = 0; r < 8; ++r) {
      int row = q0 + w * 16 + 8 * g + r;
      out[(size_t)row * DS + h * 128 + dc * 16 + ln] = ao[dc][r] / li[r];
    }
}

// ---------------------------------------------------------------------------
// img residual: out = attn_img + layer_scale * ip_out   (IP_SCALE = 1)
// ---------------------------------------------------------------------------
__global__ __launch_bounds__(256) void add_scale_kernel(
    const float* __restrict__ a, const float* __restrict__ b,
    const float* __restrict__ ls, float* __restrict__ out, int n)
{
  int i = blockIdx.x * blockDim.x + threadIdx.x;
  if (i < n) out[i] = a[i] + ls[0] * b[i];
}

// ---------------------------------------------------------------------------
extern "C" void kernel_launch(void* const* d_in, const int* in_sizes, int n_in,
                              void* d_out, int out_size, void* d_ws, size_t ws_size,
                              hipStream_t stream)
{
  const int SIMG = 2048, STXT = 512, STOT = 2560, D = 3072, H = 24;

  const float* hidden = (const float*)d_in[0];
  const float* enc    = (const float*)d_in[1];
  const float* ipenc  = (const float*)d_in[2];
  const float* lscale = (const float*)d_in[3];
  const float* Wq  = (const float*)d_in[4];   const float* bq  = (const float*)d_in[5];
  const float* Wk  = (const float*)d_in[6];   const float* bk  = (const float*)d_in[7];
  const float* Wv  = (const float*)d_in[8];   const float* bv  = (const float*)d_in[9];
  const float* nqw = (const float*)d_in[10];
  const float* nkw = (const float*)d_in[11];
  const float* Waq = (const float*)d_in[12];  const float* baq = (const float*)d_in[13];
  const float* Wak = (const float*)d_in[14];  const float* bak = (const float*)d_in[15];
  const float* Wav = (const float*)d_in[16];  const float* bav = (const float*)d_in[17];
  const float* naqw = (const float*)d_in[18];
  const float* nakw = (const float*)d_in[19];
  const float* Wkip = (const float*)d_in[20];
  const float* Wvip = (const float*)d_in[21];
  const float* Wo  = (const float*)d_in[22];  const float* bo  = (const float*)d_in[23];
  const float* Wao = (const float*)d_in[24];  const float* bao = (const float*)d_in[25];
  const float* cosb = (const float*)d_in[26];
  const float* sinb = (const float*)d_in[27];

  // workspace carve-up (floats); total ~68M floats (~271 MB)
  float* ws = (float*)d_ws;
  size_t off = 0;
  auto alloc = [&](size_t n) { float* p = ws + off; off += n; return p; };
  float* qbuf  = alloc((size_t)SIMG * D);
  float* kbuf  = alloc((size_t)SIMG * D);
  float* vbuf  = alloc((size_t)SIMG * D);
  float* eq    = alloc((size_t)STXT * D);
  float* ek    = alloc((size_t)STXT * D);
  float* ev    = alloc((size_t)STXT * D);
  float* ipk   = alloc((size_t)4 * D);
  float* ipv   = alloc((size_t)4 * D);
  float* ipout = alloc((size_t)SIMG * D);
  float* qcat  = alloc((size_t)STOT * D);
  float* kcat  = alloc((size_t)STOT * D);
  float* vcat  = alloc((size_t)STOT * D);
  float* attn  = alloc((size_t)STOT * D);
  float* imgt  = alloc((size_t)SIMG * D);

  dim3 blk256(256);

  // 1) projections (tile 128x128)
  dim3 gImg(D / 128, SIMG / 128);
  gemm_bias_wmma<<<gImg, blk256, 0, stream>>>(hidden, Wq, bq, qbuf, SIMG, D, D);
  gemm_bias_wmma<<<gImg, blk256, 0, stream>>>(hidden, Wk, bk, kbuf, SIMG, D, D);
  gemm_bias_wmma<<<gImg, blk256, 0, stream>>>(hidden, Wv, bv, vbuf, SIMG, D, D);
  dim3 gTxt(D / 128, STXT / 128);
  gemm_bias_wmma<<<gTxt, blk256, 0, stream>>>(enc, Waq, baq, eq, STXT, D, D);
  gemm_bias_wmma<<<gTxt, blk256, 0, stream>>>(enc, Wak, bak, ek, STXT, D, D);
  gemm_bias_wmma<<<gTxt, blk256, 0, stream>>>(enc, Wav, bav, ev, STXT, D, D);
  dim3 gIp(D / 128, 1);
  gemm_bias_wmma<<<gIp, blk256, 0, stream>>>(ipenc, Wkip, nullptr, ipk, 4, D, D);
  gemm_bias_wmma<<<gIp, blk256, 0, stream>>>(ipenc, Wvip, nullptr, ipv, 4, D, D);

  // 2) RMSNorms (rows = tokens*heads, 128 dims each)
  int rImg = SIMG * H, rTxt = STXT * H, rIp = 4 * H;
  rmsnorm_kernel<<<(rImg + 7) / 8, blk256, 0, stream>>>(qbuf, nqw, 1e-6f, rImg);
  rmsnorm_kernel<<<(rImg + 7) / 8, blk256, 0, stream>>>(kbuf, nkw, 1e-6f, rImg);
  rmsnorm_kernel<<<(rTxt + 7) / 8, blk256, 0, stream>>>(eq, naqw, 1e-6f, rTxt);
  rmsnorm_kernel<<<(rTxt + 7) / 8, blk256, 0, stream>>>(ek, nakw, 1e-6f, rTxt);
  rmsnorm_kernel<<<(rIp + 7) / 8, blk256, 0, stream>>>(ipk, nullptr, 1e-5f, rIp);

  // 3) IP attention (q post-rmsnorm, pre-rope)
  ip_attn_kernel<<<(rImg + 7) / 8, blk256, 0, stream>>>(qbuf, ipk, ipv, ipout);

  // 4) RoPE + concat
  int pairBlocks = (STOT * (D / 2) + 255) / 256;
  rope_concat_kernel<<<pairBlocks, blk256, 0, stream>>>(eq, qbuf, cosb, sinb, qcat, 1);
  rope_concat_kernel<<<pairBlocks, blk256, 0, stream>>>(ek, kbuf, cosb, sinb, kcat, 1);
  rope_concat_kernel<<<pairBlocks, blk256, 0, stream>>>(ev, vbuf, cosb, sinb, vcat, 0);

  // 5) main flash attention: 24 heads x 20 query blocks of 128
  flash_attn_wmma<<<H * (STOT / 128), blk256, 0, stream>>>(qcat, kcat, vcat, attn);

  // 6) img residual: attn[512:] + ls * ip_out
  int nImg = SIMG * D;
  add_scale_kernel<<<(nImg + 255) / 256, blk256, 0, stream>>>(
      attn + (size_t)STXT * D, ipout, lscale, imgt, nImg);

  // 7) output projections -> d_out (img_out then enc_out)
  float* outp = (float*)d_out;
  gemm_bias_wmma<<<gImg, blk256, 0, stream>>>(imgt, Wo, bo, outp, SIMG, D, D);
  gemm_bias_wmma<<<gTxt, blk256, 0, stream>>>(attn, Wao, bao,
                                              outp + (size_t)SIMG * D, STXT, D, D);
}
